// GATLayer_35940286333161
// MI455X (gfx1250) — compile-verified, matrix operand order checked by reading, not run on hardware
//
#include <hip/hip_runtime.h>
#include <stdint.h>

typedef __attribute__((ext_vector_type(16))) __bf16 v16bf;
typedef __attribute__((ext_vector_type(8)))  __bf16 v8bf;
typedef __attribute__((ext_vector_type(8)))  float  v8f;

#define GAT_B 1024
#define GAT_N 75
#define GAT_F 512
#define GAT_F1 256
#define GAT_M (GAT_B * GAT_N)   // 76800 = 600 * 128
#define NEG_BIAS (-1.0e9f)
#define ALPHA_SLOPE 0.2f

// round-to-nearest-even float -> bf16 (bit domain only; no __bf16 arithmetic)
__device__ __forceinline__ uint16_t f2bf(float f) {
  uint32_t u = __builtin_bit_cast(uint32_t, f);
  u += 0x7FFFu + ((u >> 16) & 1u);
  return (uint16_t)(u >> 16);
}

__device__ __forceinline__ float wred_sum(float v) {
#pragma unroll
  for (int o = 16; o > 0; o >>= 1) v += __shfl_xor(v, o, 32);
  return v;
}
__device__ __forceinline__ float wred_max(float v) {
#pragma unroll
  for (int o = 16; o > 0; o >>= 1) v = fmaxf(v, __shfl_xor(v, o, 32));
  return v;
}

// ---------------------------------------------------------------------------
// Kernel A: h = X @ W  (bf16 WMMA, f32 accumulate), written to out[:, :, F1:2F1]
// Block tile 128x128, BK=32. 256 threads = 8 waves; each wave owns 2x4 tiles.
// ---------------------------------------------------------------------------
#define APITCH 40   // halfwords; 80B row stride (16B aligned, conflict-skewed)
#define BPITCH 40

__global__ __launch_bounds__(256)
void gat_gemm_h(const float* __restrict__ X, const float* __restrict__ W,
                float* __restrict__ out) {
  __shared__ uint16_t Asb[128 * APITCH];
  __shared__ uint16_t Bsb[128 * BPITCH];

  const int tid  = threadIdx.x;
  const int lane = tid & 31;
  const int wave = tid >> 5;
  const int row0 = blockIdx.x * 128;
  const int col0 = blockIdx.y * 128;

  const int mt0   = (wave >> 1) * 32;   // wave's M offset within block tile
  const int nt0   = (wave & 1) * 64;    // wave's N offset within block tile
  const int mrow  = lane & 15;          // A/B fragment: lane -> row/col
  const int khalf = (lane >> 4) * 8;    // K half selected by lane group

  v8f acc[2][4];
#pragma unroll
  for (int i = 0; i < 2; ++i)
#pragma unroll
    for (int j = 0; j < 4; ++j)
#pragma unroll
      for (int e = 0; e < 8; ++e) acc[i][j][e] = 0.0f;

  for (int kt = 0; kt < GAT_F / 32; ++kt) {
    // --- stage X tile (128x32) as bf16, row-major ---
#pragma unroll
    for (int p = 0; p < 4; ++p) {
      int idx = tid + 256 * p;              // 0..1023
      int r   = idx >> 3;
      int c4  = idx & 7;
      const float4 v = *(const float4*)(X + (size_t)(row0 + r) * GAT_F + kt * 32 + c4 * 4);
      uint2 pk;
      pk.x = (uint32_t)f2bf(v.x) | ((uint32_t)f2bf(v.y) << 16);
      pk.y = (uint32_t)f2bf(v.z) | ((uint32_t)f2bf(v.w) << 16);
      *(uint2*)&Asb[r * APITCH + c4 * 4] = pk;
    }
    // --- stage W tile (32x128) as bf16, TRANSPOSED -> Bt[n][k] ---
#pragma unroll
    for (int p = 0; p < 4; ++p) {
      int idx = tid + 256 * p;
      int k   = idx >> 5;
      int n4  = idx & 31;
      const float4 v = *(const float4*)(W + (size_t)(kt * 32 + k) * GAT_F1 + col0 + n4 * 4);
      Bsb[(n4 * 4 + 0) * BPITCH + k] = f2bf(v.x);
      Bsb[(n4 * 4 + 1) * BPITCH + k] = f2bf(v.y);
      Bsb[(n4 * 4 + 2) * BPITCH + k] = f2bf(v.z);
      Bsb[(n4 * 4 + 3) * BPITCH + k] = f2bf(v.w);
    }
    __syncthreads();

    // --- load fragments (16-bit A-matrix layout: lane=M, K halves by lane/16) ---
    v16bf afr[2], bfr[4];
#pragma unroll
    for (int i = 0; i < 2; ++i) {
      const int rb = (mt0 + i * 16 + mrow) * APITCH + khalf;
      v8bf lo = *(const v8bf*)&Asb[rb];
      v8bf hi = *(const v8bf*)&Asb[rb + 16];
#pragma unroll
      for (int e = 0; e < 8; ++e) { afr[i][e] = lo[e]; afr[i][e + 8] = hi[e]; }
    }
#pragma unroll
    for (int j = 0; j < 4; ++j) {
      const int rb = (nt0 + j * 16 + mrow) * BPITCH + khalf;
      v8bf lo = *(const v8bf*)&Bsb[rb];
      v8bf hi = *(const v8bf*)&Bsb[rb + 16];
#pragma unroll
      for (int e = 0; e < 8; ++e) { bfr[j][e] = lo[e]; bfr[j][e + 8] = hi[e]; }
    }

#pragma unroll
    for (int i = 0; i < 2; ++i)
#pragma unroll
      for (int j = 0; j < 4; ++j)
        acc[i][j] = __builtin_amdgcn_wmma_f32_16x16x32_bf16(
            false, afr[i], false, bfr[j], (short)0, acc[i][j], false, false);
    __syncthreads();
  }

  // epilogue: C layout — lane<16: M=r, N=lane; lane>=16: M=8+r, N=lane-16
  const int mbase = (lane >> 4) * 8;
  const int ncol  = lane & 15;
#pragma unroll
  for (int i = 0; i < 2; ++i)
#pragma unroll
    for (int j = 0; j < 4; ++j)
#pragma unroll
      for (int r = 0; r < 8; ++r) {
        int m = mt0 + i * 16 + mbase + r;
        int n = nt0 + j * 16 + ncol;
        out[(size_t)(row0 + m) * (2 * GAT_F1) + GAT_F1 + col0 + n] = acc[i][j][r];
      }
}

// ---------------------------------------------------------------------------
// Kernel B: s0[r] = h[r]·a0, s1[r] = h[r]·a1  (one wave per row, fp32)
// ---------------------------------------------------------------------------
__global__ __launch_bounds__(256)
void gat_scores(const float* __restrict__ out, const float* __restrict__ avec,
                float* __restrict__ s0w, float* __restrict__ s1w) {
  const int lane = threadIdx.x & 31;
  const int wave = threadIdx.x >> 5;
  const size_t row = (size_t)blockIdx.x * 8 + wave;
  const float* h  = out + row * (2 * GAT_F1) + GAT_F1;
  const float* a0 = avec;
  const float* a1 = avec + GAT_F1;
  float v0 = 0.f, v1 = 0.f;
#pragma unroll
  for (int s = 0; s < GAT_F1 / 32; ++s) {
    int o = lane + 32 * s;
    float hv = h[o];
    v0 = fmaf(hv, a0[o], v0);
    v1 = fmaf(hv, a1[o], v1);
  }
  v0 = wred_sum(v0);
  v1 = wred_sum(v1);
  if (lane == 0) { s0w[row] = v0; s1w[row] = v1; }
}

// ---------------------------------------------------------------------------
// Kernel C: per batch — softmax(leakyrelu(s0_i+s1_j) + mask) @ h via WMMA.
// alpha padded to 80x96 bf16 (zeros), h^T staged 256x96 bf16 in LDS.
// ---------------------------------------------------------------------------
#define CP 96   // K pitch in halfwords (192B rows, 16B aligned)

__global__ __launch_bounds__(256)
void gat_softmax_agg(const float* __restrict__ out_h, const int* __restrict__ Adj,
                     const float* __restrict__ s0w, const float* __restrict__ s1w,
                     float* __restrict__ out) {
  __shared__ uint16_t alphas[80 * CP];       // 15360 B
  __shared__ uint16_t hTs[GAT_F1 * CP];      // 49152 B
  __shared__ float s0s[80];
  __shared__ float s1s[80];

  const int tid  = threadIdx.x;
  const int lane = tid & 31;
  const int wave = tid >> 5;
  const int b    = blockIdx.x;
  const size_t rbase = (size_t)b * GAT_N;

  // phase 1: zero-pad LDS, load scores
  for (int i = tid; i < 80 * CP / 2; i += 256)      ((uint32_t*)alphas)[i] = 0u;
  for (int i = tid; i < GAT_F1 * CP / 2; i += 256)  ((uint32_t*)hTs)[i]    = 0u;
  if (tid < GAT_N) { s0s[tid] = s0w[rbase + tid]; s1s[tid] = s1w[rbase + tid]; }
  __syncthreads();

  // phase 2a: stage h^T as bf16  (hTs[feature][node])
  for (int idx = tid; idx < GAT_N * (GAT_F1 / 4); idx += 256) {
    int j  = idx >> 6;   // node
    int f4 = idx & 63;   // feature group of 4
    const float4 v = *(const float4*)(out_h + (rbase + j) * (2 * GAT_F1) + GAT_F1 + f4 * 4);
    hTs[(f4 * 4 + 0) * CP + j] = f2bf(v.x);
    hTs[(f4 * 4 + 1) * CP + j] = f2bf(v.y);
    hTs[(f4 * 4 + 2) * CP + j] = f2bf(v.z);
    hTs[(f4 * 4 + 3) * CP + j] = f2bf(v.w);
  }

  // phase 2b: fp32 masked softmax, one row per wave (3 j-slots per lane)
  for (int i = wave; i < GAT_N; i += 8) {
    const float si = s0s[i];
    float ev[3];
    float mx = -3.0e38f;
#pragma unroll
    for (int s = 0; s < 3; ++s) {
      int j = lane + 32 * s;
      float e = -3.0e38f;
      if (j < GAT_N) {
        float x  = si + s1s[j];
        float lr = (x > 0.f) ? x : ALPHA_SLOPE * x;
        e = lr + ((Adj[i * GAT_N + j] != 0) ? 0.f : NEG_BIAS);
      }
      ev[s] = e;
      mx = fmaxf(mx, e);
    }
    mx = wred_max(mx);
    float p[3];
    float sum = 0.f;
#pragma unroll
    for (int s = 0; s < 3; ++s) {
      int j = lane + 32 * s;
      p[s] = (j < GAT_N) ? __expf(ev[s] - mx) : 0.f;
      sum += p[s];
    }
    sum = wred_sum(sum);
    const float inv = 1.0f / sum;
#pragma unroll
    for (int s = 0; s < 3; ++s) {
      int j = lane + 32 * s;
      if (j < GAT_N) alphas[i * CP + j] = f2bf(p[s] * inv);
    }
  }
  __syncthreads();

  // phase 3: out[b] = alpha @ h  — 5x16 output tiles, K=96 (3 WMMAs each)
  const int mrow  = lane & 15;
  const int khalf = (lane >> 4) * 8;
  const int mbase = (lane >> 4) * 8;
  const int ncol  = lane & 15;
  for (int t = wave; t < 80; t += 8) {        // uniform: 10 iters per wave
    const int mt = t >> 4;
    const int nt = t & 15;
    v8f acc;
#pragma unroll
    for (int e = 0; e < 8; ++e) acc[e] = 0.f;
#pragma unroll
    for (int kt = 0; kt < 3; ++kt) {
      const int kb = kt * 32 + khalf;
      v8bf alo = *(const v8bf*)&alphas[(mt * 16 + mrow) * CP + kb];
      v8bf ahi = *(const v8bf*)&alphas[(mt * 16 + mrow) * CP + kb + 16];
      v8bf blo = *(const v8bf*)&hTs[(nt * 16 + mrow) * CP + kb];
      v8bf bhi = *(const v8bf*)&hTs[(nt * 16 + mrow) * CP + kb + 16];
      v16bf afv, bfv;
#pragma unroll
      for (int e = 0; e < 8; ++e) {
        afv[e] = alo[e]; afv[e + 8] = ahi[e];
        bfv[e] = blo[e]; bfv[e + 8] = bhi[e];
      }
      acc = __builtin_amdgcn_wmma_f32_16x16x32_bf16(
          false, afv, false, bfv, (short)0, acc, false, false);
    }
#pragma unroll
    for (int r = 0; r < 8; ++r) {
      int m = mt * 16 + mbase + r;
      if (m < GAT_N)
        out[(rbase + m) * (2 * GAT_F1) + nt * 16 + ncol] = acc[r];
    }
  }
}

// ---------------------------------------------------------------------------
extern "C" void kernel_launch(void* const* d_in, const int* in_sizes, int n_in,
                              void* d_out, int out_size, void* d_ws, size_t ws_size,
                              hipStream_t stream) {
  (void)in_sizes; (void)n_in; (void)out_size; (void)ws_size;
  const float* X = (const float*)d_in[0];
  const int*   A = (const int*)d_in[1];
  const float* W = (const float*)d_in[2];
  const float* a = (const float*)d_in[3];
  float* out = (float*)d_out;
  float* s0w = (float*)d_ws;            // 76800 floats
  float* s1w = s0w + GAT_M;             // 76800 floats

  dim3 gA(GAT_M / 128, GAT_F1 / 128);   // 600 x 2
  gat_gemm_h<<<gA, 256, 0, stream>>>(X, W, out);
  gat_scores<<<GAT_M / 8, 256, 0, stream>>>(out, a, s0w, s1w);
  gat_softmax_agg<<<GAT_B, 256, 0, stream>>>(out, A, s0w, s1w, out);
}
